// GNN_63385127354906
// MI455X (gfx1250) — matrix operand order repaired and checked by reading, not compiled, not run on hardware
//
#include <hip/hip_runtime.h>
#include <hip/hip_bf16.h>

typedef __attribute__((ext_vector_type(16))) _Float16 v16h;
typedef __attribute__((ext_vector_type(8)))  _Float16 v8h;
typedef __attribute__((ext_vector_type(8)))  float    v8f;

#define TPB 256

// ---------- utility kernels ----------
__global__ void k_fill(float* __restrict__ p, float v, long n) {
    long i = (long)blockIdx.x * blockDim.x + threadIdx.x;
    long stride = (long)gridDim.x * blockDim.x;
    for (; i < n; i += stride) p[i] = v;
}

__global__ void k_cvt_f16(const float* __restrict__ src, _Float16* __restrict__ dst, long n) {
    long i = (long)blockIdx.x * blockDim.x + threadIdx.x;
    if (i < n) dst[i] = (_Float16)src[i];
}

__device__ __forceinline__ void atomicMaxF(float* addr, float val) {
    int* ai = (int*)addr;
    int old = __float_as_int(*addr);
    while (__int_as_float(old) < val) {
        int prev = atomicCAS(ai, old, __float_as_int(val));
        if (prev == old) break;
        old = prev;
    }
}

// ---------- degree ----------
__global__ void k_deg(const int* __restrict__ col, const float* __restrict__ w,
                      float* __restrict__ deg, int E) {
    int e = blockIdx.x * blockDim.x + threadIdx.x;
    if (e < E) atomicAdd(&deg[col[e]], w[e]);
}

__global__ void k_dinv(const float* __restrict__ deg, float* __restrict__ dinv, int n) {
    int i = blockIdx.x * blockDim.x + threadIdx.x;
    if (i < n) dinv[i] = rsqrtf(deg[i]);
}

// ---------- WMMA GEMM: C[M,NOUT] = A16[M,128] * W16[NOUT,128]^T (+bias) ----------
// Strip-mined: one wave owns a 16-row x NOUT output strip.  A fragments for all
// 4 k-steps are held in VGPRs; the weight panel is staged once per block into
// LDS and B fragments come from ds_load.  M must be a multiple of 16.
template <int NOUT>
__global__ void k_gemm_wmma(const _Float16* __restrict__ A16,
                            const _Float16* __restrict__ W16,
                            const float* __restrict__ bias,
                            float* __restrict__ C, int M) {
    __shared__ _Float16 sB[NOUT * 128];
    // cooperative stage of the whole weight panel (NOUT x 128 f16)
    for (int idx = threadIdx.x * 8; idx < NOUT * 128; idx += TPB * 8) {
        *(v8h*)(sB + idx) = *(const v8h*)(W16 + idx);
    }
    __syncthreads();

    int wave = threadIdx.x >> 5;
    int lane = threadIdx.x & 31;
    int strip = blockIdx.x * (TPB >> 5) + wave;       // 16-row strip index
    if (strip >= (M >> 4)) return;                    // wave-uniform exit
    int tm  = strip << 4;
    int l15 = lane & 31 & 15;
    int hi  = lane >> 4;                              // 0: lanes 0-15, 1: lanes 16-31

    // preload A fragments for all 4 k-steps (contiguous k = 16*hi + s, matching B)
    const _Float16* arow = A16 + (long)(tm + l15) * 128;
    v16h a[4];
    #pragma unroll
    for (int ks = 0; ks < 4; ++ks) {
        a[ks] = *(const v16h*)(arow + ks * 32 + hi * 16);
    }

    #pragma unroll
    for (int tn = 0; tn < NOUT; tn += 16) {
        v8f acc = {};
        const _Float16* brow = sB + (tn + l15) * 128 + hi * 16;
        #pragma unroll
        for (int ks = 0; ks < 4; ++ks) {
            v16h b = *(const v16h*)(brow + ks * 32);  // 32B-aligned LDS load
            acc = __builtin_amdgcn_wmma_f32_16x16x32_f16(false, a[ks], false, b,
                                                         (short)0, acc, false, false);
        }
        int colx = tn + l15;
        float bv = bias ? bias[colx] : 0.0f;
        #pragma unroll
        for (int r = 0; r < 8; ++r)                   // VGPR r: M = tm + r + hi*8
            C[(long)(tm + r + hi * 8) * NOUT + colx] = acc[r] + bv;
    }
}

// ---------- GCN ----------
__global__ void k_hinit(const float* __restrict__ xl, const float* __restrict__ deg,
                        const float* __restrict__ bg, float* __restrict__ h, long n128) {
    long i = (long)blockIdx.x * blockDim.x + threadIdx.x;
    if (i >= n128) return;
    int node = (int)(i >> 7), f = (int)(i & 127);
    h[i] = xl[i] / deg[node] + bg[f];
}

__global__ void k_gcn_scatter(const int* __restrict__ row, const int* __restrict__ col,
                              const float* __restrict__ w, const float* __restrict__ dinv,
                              const float* __restrict__ xl, float* __restrict__ h, int E) {
    long gid = (long)blockIdx.x * blockDim.x + threadIdx.x;
    int e = (int)(gid >> 5), lane = (int)(gid & 31);
    if (e >= E) return;
    int r = row[e], c = col[e];
    float coef = dinv[r] * w[e] * dinv[c];
    float4 xv = ((const float4*)(xl + (long)r * 128))[lane];
    float* hp = h + (long)c * 128 + lane * 4;
    atomicAdd(hp + 0, coef * xv.x);
    atomicAdd(hp + 1, coef * xv.y);
    atomicAdd(hp + 2, coef * xv.z);
    atomicAdd(hp + 3, coef * xv.w);
}

// ---------- attention ----------
__global__ void k_alpha(const int* __restrict__ row, const int* __restrict__ col,
                        const float* __restrict__ q, const float* __restrict__ kk,
                        float* __restrict__ alpha, float* __restrict__ amax, int E) {
    long gid = (long)blockIdx.x * blockDim.x + threadIdx.x;
    int e = (int)(gid >> 5), lane = (int)(gid & 31);
    if (e >= E) return;
    int r = row[e], c = col[e];
    int hh = lane >> 3, j = lane & 7;                 // head, 4-dim chunk within head
    float4 qv = *(const float4*)(q  + (long)c * 128 + hh * 32 + j * 4);
    float4 kv = *(const float4*)(kk + (long)r * 128 + hh * 32 + j * 4);
    float p = qv.x * kv.x + qv.y * kv.y + qv.z * kv.z + qv.w * kv.w;
    p += __shfl_xor(p, 1);
    p += __shfl_xor(p, 2);
    p += __shfl_xor(p, 4);
    if (j == 0) {
        float a = p * 0.17677669529663687f;           // 1/sqrt(32)
        alpha[(long)e * 4 + hh] = a;
        atomicMaxF(&amax[(long)c * 4 + hh], a);
    }
}

__global__ void k_ex(const int* __restrict__ col, float* __restrict__ alpha_ex,
                     const float* __restrict__ amax, float* __restrict__ denom, long EH) {
    long i = (long)blockIdx.x * blockDim.x + threadIdx.x;
    if (i >= EH) return;
    int e = (int)(i >> 2), hh = (int)(i & 3);
    int c = col[e];
    float ex = expf(alpha_ex[i] - amax[(long)c * 4 + hh]);
    alpha_ex[i] = ex;
    atomicAdd(&denom[(long)c * 4 + hh], ex);
}

__global__ void k_agg(const int* __restrict__ row, const int* __restrict__ col,
                      const float* __restrict__ ex, const float* __restrict__ denom,
                      const float* __restrict__ v, float* __restrict__ agg, int E) {
    long gid = (long)blockIdx.x * blockDim.x + threadIdx.x;
    int e = (int)(gid >> 5), lane = (int)(gid & 31);
    if (e >= E) return;
    int r = row[e], c = col[e];
    int hh = lane >> 3;                               // feature block lane*4 lies in head lane>>3
    float attn = ex[(long)e * 4 + hh] / fmaxf(denom[(long)c * 4 + hh], 1e-16f);
    float4 vv = ((const float4*)(v + (long)r * 128))[lane];
    float* ap = agg + (long)c * 128 + lane * 4;
    atomicAdd(ap + 0, attn * vv.x);
    atomicAdd(ap + 1, attn * vv.y);
    atomicAdd(ap + 2, attn * vv.z);
    atomicAdd(ap + 3, attn * vv.w);
}

// ---------- head mean + BN statistics ----------
__global__ void k_out(const float* __restrict__ agg, const float* __restrict__ hs,
                      float* __restrict__ out, float* __restrict__ sums, int n) {
    int t = threadIdx.x;
    long i = (long)blockIdx.x * TPB + t;              // node-major: node = i>>5, ch = i&31
    int node = (int)(i >> 5), ch = (int)(i & 31);
    float s = 0.0f;
    if (node < n) {
        const float* a = agg + (long)node * 128;
        s = (a[ch] + a[32 + ch] + a[64 + ch] + a[96 + ch]) * 0.25f + hs[(long)node * 32 + ch];
        out[i] = s;
    }
    __shared__ float red[TPB], red2[TPB];
    red[t] = s; red2[t] = s * s;
    __syncthreads();
    for (int off = 128; off >= 32; off >>= 1) {
        if (t < off) { red[t] += red[t + off]; red2[t] += red2[t + off]; }
        __syncthreads();
    }
    if (t < 32) {
        atomicAdd(&sums[t], red[t]);
        atomicAdd(&sums[32 + t], red2[t]);
    }
}

__global__ void k_bn(const float* __restrict__ out, const float* __restrict__ sums,
                     const float* __restrict__ gamma, const float* __restrict__ beta,
                     float* __restrict__ y, long n32, float invN) {
    long i = (long)blockIdx.x * blockDim.x + threadIdx.x;
    if (i >= n32) return;
    int ch = (int)(i & 31);
    float mu  = sums[ch] * invN;
    float var = sums[32 + ch] * invN - mu * mu;
    float t = gamma[ch] * (out[i] - mu) * rsqrtf(var + 1e-5f) + beta[ch];
    y[i] = t > 0.0f ? t : 0.01f * t;
}

// ---------- launcher ----------
extern "C" void kernel_launch(void* const* d_in, const int* in_sizes, int n_in,
                              void* d_out, int out_size, void* d_ws, size_t ws_size,
                              hipStream_t stream) {
    const float* X    = (const float*)d_in[0];
    const int*   ei   = (const int*)d_in[1];
    const float* ew   = (const float*)d_in[2];
    const float* Wg   = (const float*)d_in[3];
    const float* bg   = (const float*)d_in[4];
    const float* Wq   = (const float*)d_in[5];
    const float* bq   = (const float*)d_in[6];
    const float* Wk   = (const float*)d_in[7];
    const float* bk   = (const float*)d_in[8];
    const float* Wv   = (const float*)d_in[9];
    const float* bv   = (const float*)d_in[10];
    const float* Wsm  = (const float*)d_in[11];
    const float* bs   = (const float*)d_in[12];
    const float* gam  = (const float*)d_in[13];
    const float* bet  = (const float*)d_in[14];
    float* Y = (float*)d_out;

    const int N = in_sizes[0] / 128;
    const int E = in_sizes[2];
    const int* row = ei;
    const int* col = ei + E;

    // workspace carve-out (256B aligned regions)
    char* base = (char*)d_ws;
    size_t off = 0;
    auto carve = [&](size_t bytes) { char* p = base + off; off = (off + bytes + 255) & ~(size_t)255; return p; };
    float*    deg   = (float*)carve((size_t)N * 4);
    float*    dinv  = (float*)carve((size_t)N * 4);
    float*    amax  = (float*)carve((size_t)N * 4 * 4);
    float*    denom = (float*)carve((size_t)N * 4 * 4);
    float*    sums  = (float*)carve(64 * 4);
    float*    xl    = (float*)carve((size_t)N * 128 * 4);
    float*    h     = (float*)carve((size_t)N * 128 * 4);
    float*    q     = (float*)carve((size_t)N * 128 * 4);
    float*    kbuf  = (float*)carve((size_t)N * 128 * 4);
    float*    vbuf  = (float*)carve((size_t)N * 128 * 4);
    float*    hs    = (float*)carve((size_t)N * 32 * 4);
    float*    agg   = (float*)carve((size_t)N * 128 * 4);
    float*    exb   = (float*)carve((size_t)E * 4 * 4);
    float*    outb  = (float*)carve((size_t)N * 32 * 4);
    _Float16* x16   = (_Float16*)carve((size_t)N * 128 * 2);
    _Float16* h16   = (_Float16*)carve((size_t)N * 128 * 2);
    _Float16* wg16  = (_Float16*)carve((size_t)128 * 128 * 2);
    _Float16* wq16  = (_Float16*)carve((size_t)128 * 128 * 2);
    _Float16* wk16  = (_Float16*)carve((size_t)128 * 128 * 2);
    _Float16* wv16  = (_Float16*)carve((size_t)128 * 128 * 2);
    _Float16* ws16  = (_Float16*)carve((size_t)32 * 128 * 2);

    auto cdiv = [](long a, long b) { return (int)((a + b - 1) / b); };

    // init accumulators (every call: graph replays don't re-zero)
    k_fill<<<1024, TPB, 0, stream>>>(deg, 1.0f, N);
    k_fill<<<1024, TPB, 0, stream>>>(amax, 0.0f, (long)N * 4);
    k_fill<<<1024, TPB, 0, stream>>>(denom, 0.0f, (long)N * 4);
    k_fill<<<1024, TPB, 0, stream>>>(agg, 0.0f, (long)N * 128);
    k_fill<<<1, 64, 0, stream>>>(sums, 0.0f, 64);

    // degree + normalization
    k_deg<<<cdiv(E, TPB), TPB, 0, stream>>>(col, ew, deg, E);
    k_dinv<<<cdiv(N, TPB), TPB, 0, stream>>>(deg, dinv, N);

    // f32 -> f16 conversions
    k_cvt_f16<<<cdiv((long)N * 128, TPB), TPB, 0, stream>>>(X, x16, (long)N * 128);
    k_cvt_f16<<<cdiv(128 * 128, TPB), TPB, 0, stream>>>(Wg, wg16, 128 * 128);
    k_cvt_f16<<<cdiv(128 * 128, TPB), TPB, 0, stream>>>(Wq, wq16, 128 * 128);
    k_cvt_f16<<<cdiv(128 * 128, TPB), TPB, 0, stream>>>(Wk, wk16, 128 * 128);
    k_cvt_f16<<<cdiv(128 * 128, TPB), TPB, 0, stream>>>(Wv, wv16, 128 * 128);
    k_cvt_f16<<<cdiv(32 * 128, TPB), TPB, 0, stream>>>(Wsm, ws16, 32 * 128);

    const int strips = N / 16;                 // N % 16 == 0 (50000)
    const int gemm_blocks = cdiv(strips, TPB >> 5);

    // xl = X @ Wg^T   (bias folded into k_hinit)
    k_gemm_wmma<128><<<gemm_blocks, TPB, 0, stream>>>(x16, wg16, nullptr, xl, N);

    // h = xl/deg + bg, then scattered neighbor sum
    k_hinit<<<cdiv((long)N * 128, TPB), TPB, 0, stream>>>(xl, deg, bg, h, (long)N * 128);
    k_gcn_scatter<<<cdiv((long)E * 32, TPB), TPB, 0, stream>>>(row, col, ew, dinv, xl, h, E);

    // q/k/v/hs projections from h
    k_cvt_f16<<<cdiv((long)N * 128, TPB), TPB, 0, stream>>>(h, h16, (long)N * 128);
    k_gemm_wmma<128><<<gemm_blocks, TPB, 0, stream>>>(h16, wq16, bq, q, N);
    k_gemm_wmma<128><<<gemm_blocks, TPB, 0, stream>>>(h16, wk16, bk, kbuf, N);
    k_gemm_wmma<128><<<gemm_blocks, TPB, 0, stream>>>(h16, wv16, bv, vbuf, N);
    k_gemm_wmma<32><<<gemm_blocks, TPB, 0, stream>>>(h16, ws16, bs, hs, N);

    // attention softmax over incoming edges
    k_alpha<<<cdiv((long)E * 32, TPB), TPB, 0, stream>>>(row, col, q, kbuf, exb, amax, E);
    k_ex<<<cdiv((long)E * 4, TPB), TPB, 0, stream>>>(col, exb, amax, denom, (long)E * 4);
    k_agg<<<cdiv((long)E * 32, TPB), TPB, 0, stream>>>(row, col, exb, denom, vbuf, agg, E);

    // head mean + skip, BN stats, normalize + leaky relu
    k_out<<<cdiv((long)N * 32, TPB), TPB, 0, stream>>>(agg, hs, outb, sums, N);
    k_bn<<<cdiv((long)N * 32, TPB), TPB, 0, stream>>>(outb, sums, gam, bet, Y, (long)N * 32, 1.0f / (float)N);
}
